// ModalityEnhancedLSTM_60060822667631
// MI455X (gfx1250) — matrix operand order repaired and checked by reading, not compiled, not run on hardware
//
#include <hip/hip_runtime.h>
#include <stdint.h>

// ---------------------------------------------------------------------------
// ModalityEnhancedLSTM for MI455X (gfx1250, wave32, WMMA)
//   B=64, S=1024, D=256, H=256, 4H=1024
// Phase 1: x_proj = x @ W + bias  (memory-bound GEMM, bf16x3 WMMA, W pre-packed)
// Phase 2: sequential LSTM scan, batch-partitioned across 4 workgroups.
//   - U pre-packed in per-lane WMMA B-fragment layout (streamed from L2)
//   - h kept in LDS as pre-packed bf16 hi/lo pairs (A-fragment layout),
//     double-buffered -> one barrier/step, zero conversion VALU on the chain
//   - x_proj gate loads issued before the GEMM (latency hidden by 96 WMMAs)
// ---------------------------------------------------------------------------

#define B_   64
#define S_   1024
#define D_   256
#define H_   256
#define G4H_ 1024

typedef uint32_t u32;
typedef __bf16 bf16;
typedef __attribute__((ext_vector_type(16))) bf16  v16bf;
typedef __attribute__((ext_vector_type(8)))  float v8f;
typedef __attribute__((ext_vector_type(4)))  u32   v4u;

// fragment union: 8 dwords == 2 x b128 == one 16-element bf16 WMMA operand
union VF { u32 u[8]; v4u q[2]; v16bf v; };

// workspace layout (in 4-byte elements)
static const size_t XP_ELEMS     = (size_t)S_ * B_ * G4H_;   // 67,108,864 f32
static const size_t UPACK_DWORDS = (size_t)64 * 8 * 32 * 8;  // 131,072 dwords per array

__device__ __forceinline__ v8f wmma_bf16(v16bf a, v16bf b, v8f c) {
  return __builtin_amdgcn_wmma_f32_16x16x32_bf16(
      false, a, false, b, (short)0, c, false, false);
}

__device__ __forceinline__ void split_bf16(float f, bf16& hi, bf16& lo) {
  hi = (bf16)f;
  lo = (bf16)(f - (float)hi);
}

union BF2 { u32 u; bf16 b[2]; };

// ---------------------------------------------------------------------------
// Kernel 1: pack a [256,1024] f32 matrix into bf16 hi/lo arrays in per-lane
// WMMA B-fragment order: index = ((ntile*8 + kstep)*32 + lane)*8 + v
//   pair: K = kstep*32 + (v>>2)*16 + (lane>>4)*8 + 2(v&3) + {0,1}, N = ntile*16 + (lane&15)
// Used for both U and W (identical shapes).
// ---------------------------------------------------------------------------
__global__ void __launch_bounds__(256) pack_bmat_kernel(const float* __restrict__ src,
                                                        u32* __restrict__ dhi,
                                                        u32* __restrict__ dlo) {
  int gid = blockIdx.x * blockDim.x + threadIdx.x;  // 0 .. 64*8*32-1
  if (gid >= 64 * 8 * 32) return;
  int lane  = gid & 31;
  int kstep = (gid >> 5) & 7;
  int ntile = gid >> 8;
  int col   = ntile * 16 + (lane & 15);
  int khalf = (lane >> 4) * 8;
#pragma unroll
  for (int v = 0; v < 8; ++v) {
    int k0 = kstep * 32 + (v >> 2) * 16 + khalf + 2 * (v & 3);
    float e0 = src[(size_t)k0 * G4H_ + col];
    float e1 = src[(size_t)(k0 + 1) * G4H_ + col];
    bf16 h0, l0, h1, l1;
    split_bf16(e0, h0, l0);
    split_bf16(e1, h1, l1);
    BF2 ph; ph.b[0] = h0; ph.b[1] = h1;
    BF2 pl; pl.b[0] = l0; pl.b[1] = l1;
    dhi[(size_t)gid * 8 + v] = ph.u;
    dlo[(size_t)gid * 8 + v] = pl.u;
  }
}

// ---------------------------------------------------------------------------
// Kernel 2: x_proj[s][b][g] = sum_d x[b][s][d]*W[d][g] + bias[g]
// grid = (S, B/16); block = 256 (8 waves). Wave w sweeps N-tiles ng*8+w.
// A (x) converted once per wave; B (W) pre-packed -> pure b128 loads.
// ---------------------------------------------------------------------------
__global__ void __launch_bounds__(256) xproj_kernel(const float* __restrict__ x,
                                                    const u32* __restrict__ whi,
                                                    const u32* __restrict__ wlo,
                                                    const float* __restrict__ bias,
                                                    float* __restrict__ xp) {
  const int s    = blockIdx.x;
  const int b0   = blockIdx.y * 16;
  const int tid  = threadIdx.x;
  const int w    = tid >> 5;
  const int lane = tid & 31;
  const int row  = lane & 15;
  const int half = lane >> 4;

  // A fragments: 16x32 bf16 per k-step, hi + lo residual (register resident)
  v16bf ahi[8], alo[8];
  const float* xrow = x + ((size_t)(b0 + row) * S_ + s) * D_;
#pragma unroll
  for (int ks = 0; ks < 8; ++ks) {
#pragma unroll
    for (int v = 0; v < 8; ++v) {
#pragma unroll
      for (int p = 0; p < 2; ++p) {
        int K = ks * 32 + (v >> 2) * 16 + half * 8 + (v & 3) * 2 + p;
        bf16 hi, lo;
        split_bf16(xrow[K], hi, lo);
        ahi[ks][2 * v + p] = hi;
        alo[ks][2 * v + p] = lo;
      }
    }
  }

  for (int ng = 0; ng < 8; ++ng) {
    const int ntile = ng * 8 + w;
    v8f acc = {};
#pragma unroll
    for (int ks = 0; ks < 8; ++ks) {
      const size_t base = ((size_t)(ntile * 8 + ks) * 32 + lane) * 8;
      VF Bh, Bl;
      Bh.q[0] = *(const v4u*)(whi + base);
      Bh.q[1] = *(const v4u*)(whi + base + 4);
      Bl.q[0] = *(const v4u*)(wlo + base);
      Bl.q[1] = *(const v4u*)(wlo + base + 4);
      acc = wmma_bf16(ahi[ks], Bh.v, acc);  // hi*hi
      acc = wmma_bf16(ahi[ks], Bl.v, acc);  // hi*lo
      acc = wmma_bf16(alo[ks], Bh.v, acc);  // lo*hi
    }
#pragma unroll
    for (int r = 0; r < 8; ++r) {
      int M = r + 8 * half;
      int N = ntile * 16 + (lane & 15);
      xp[((size_t)s * B_ + (b0 + M)) * G4H_ + N] = acc[r] + bias[N];
    }
  }
}

// ---------------------------------------------------------------------------
// Kernel 3: sequential LSTM scan. grid = 4 (16 batch rows each, independent),
// block = 512 (16 waves). Wave w owns hidden slice [w*16, w*16+16).
// h stored in LDS as packed bf16 hi/lo pairs in A-fragment dword layout,
// double-buffered. Row stride 132 dwords: 16B-aligned rows, conflict-free
// b128 reads (bank offset 4r per row).
// ---------------------------------------------------------------------------
#define HROW_ 132  // 128 dwords of packed bf16 pairs + 4 pad

__global__ void __launch_bounds__(512) lstm_scan_kernel(const float* __restrict__ xp,
                                                        const u32* __restrict__ uhi,
                                                        const u32* __restrict__ ulo,
                                                        const float* __restrict__ lam,
                                                        float* __restrict__ out) {
  __shared__ __align__(16) u32 hhi[2][16][HROW_];
  __shared__ __align__(16) u32 hlo[2][16][HROW_];

  const int b0   = blockIdx.x * 16;
  const int tid  = threadIdx.x;
  const int w    = tid >> 5;     // 0..15 hidden slice owner
  const int lane = tid & 31;
  const int row  = lane & 15;
  const int half = lane >> 4;
  const int j    = w * 16 + (lane & 15);  // hidden unit (C-layout column)

  v8f c_st = {};    // persistent cell state fragment
  float hfin[8];    // last h values (exact f32) for h_t output

  for (int idx = tid; idx < 2 * 16 * HROW_; idx += 512) {
    (&hhi[0][0][0])[idx] = 0u;
    (&hlo[0][0][0])[idx] = 0u;
  }
  __syncthreads();

  for (int s = 0; s < S_; ++s) {
    const int cur = s & 1;
    const int nxt = cur ^ 1;
    const float lam_s = lam[s];

    // --- (1) issue x_proj gate loads NOW; GEMM below hides their latency ---
    float xg[4][8];
#pragma unroll
    for (int r = 0; r < 8; ++r) {
      const int brow = b0 + r + 8 * half;
      const size_t xb = ((size_t)s * B_ + brow) * (size_t)G4H_ + j;
#pragma unroll
      for (int q = 0; q < 4; ++q) xg[q][r] = xp[xb + q * 256];
    }
    // prefetch next step's gate regions into L2 (global_prefetch_b8)
    if (s + 1 < S_) {
#pragma unroll
      for (int r = 0; r < 8; ++r) {
        const int brow = b0 + r + 8 * half;
        const size_t xb = ((size_t)(s + 1) * B_ + brow) * (size_t)G4H_ + j;
#pragma unroll
        for (int q = 0; q < 4; ++q) __builtin_prefetch(&xp[xb + q * 256], 0, 1);
      }
    }

    // --- (2) gates = h(s-1) @ U : four 16x16 tiles, bf16x3, ks-outer ---
    v8f acc[4];
    { v8f z = {}; acc[0] = z; acc[1] = z; acc[2] = z; acc[3] = z; }
    const u32* rh = &hhi[cur][row][0];
    const u32* rl = &hlo[cur][row][0];
#pragma unroll
    for (int ks = 0; ks < 8; ++ks) {
      // A fragment: pure b128 LDS loads of pre-packed bf16 pairs
      VF Ah, Al;
      const int k0 = ks * 16 + half * 4;
      Ah.q[0] = *(const v4u*)(rh + k0);
      Ah.q[1] = *(const v4u*)(rh + k0 + 8);
      Al.q[0] = *(const v4u*)(rl + k0);
      Al.q[1] = *(const v4u*)(rl + k0 + 8);
      // B fragments for all 4 gate tiles (L2-resident packed U)
      VF Bh[4], Bl[4];
#pragma unroll
      for (int q = 0; q < 4; ++q) {
        const size_t base = ((size_t)((w + 16 * q) * 8 + ks) * 32 + lane) * 8;
        Bh[q].q[0] = *(const v4u*)(uhi + base);
        Bh[q].q[1] = *(const v4u*)(uhi + base + 4);
        Bl[q].q[0] = *(const v4u*)(ulo + base);
        Bl[q].q[1] = *(const v4u*)(ulo + base + 4);
      }
      // 12 WMMAs: 4 independent accumulator chains interleaved
#pragma unroll
      for (int q = 0; q < 4; ++q) acc[q] = wmma_bf16(Ah.v, Bh[q].v, acc[q]);
#pragma unroll
      for (int q = 0; q < 4; ++q) acc[q] = wmma_bf16(Ah.v, Bl[q].v, acc[q]);
#pragma unroll
      for (int q = 0; q < 4; ++q) acc[q] = wmma_bf16(Al.v, Bh[q].v, acc[q]);
    }

    // --- (3) gate activations + state update; write h(s) packed to buf nxt ---
#pragma unroll
    for (int r = 0; r < 8; ++r) {
      const int M    = r + 8 * half;
      const int brow = b0 + M;
      float gi = xg[0][r] + acc[0][r];
      float gf = xg[1][r] + acc[1][r];
      float gg = xg[2][r] + acc[2][r];
      float go = xg[3][r] + acc[3][r];
      float i_t = 1.0f / (1.0f + __expf(-gi));
      float f_t = 1.0f / (1.0f + __expf(-gf));
      float eg  = __expf(2.0f * gg);
      float g_t = (eg - 1.0f) / (eg + 1.0f);
      float o_t = 1.0f / (1.0f + __expf(-go));
      float c   = f_t * c_st[r] + i_t * g_t * lam_s;
      c_st[r]   = c;
      float ec  = __expf(2.0f * c);
      float h   = o_t * (ec - 1.0f) / (ec + 1.0f);
      hfin[r]   = h;
      bf16 hh, hl;
      split_bf16(h, hh, hl);                  // converted ONCE, read 16x
      ((bf16*)hhi[nxt][M])[j] = hh;           // ds_store_b16
      ((bf16*)hlo[nxt][M])[j] = hl;
      out[(size_t)brow * (S_ * H_) + (size_t)s * H_ + j] = h;  // exact f32
    }
    __syncthreads();  // single barrier per step (double-buffered h)
  }

  // --- final h_t, c_t (exact f32 from registers) ---
  float* h_out = out + (size_t)B_ * S_ * H_;
  float* c_out = h_out + (size_t)B_ * H_;
#pragma unroll
  for (int r = 0; r < 8; ++r) {
    const int brow = b0 + r + 8 * half;
    h_out[(size_t)brow * H_ + j] = hfin[r];
    c_out[(size_t)brow * H_ + j] = c_st[r];
  }
}

// ---------------------------------------------------------------------------
extern "C" void kernel_launch(void* const* d_in, const int* in_sizes, int n_in,
                              void* d_out, int out_size, void* d_ws, size_t ws_size,
                              hipStream_t stream) {
  const float* x    = (const float*)d_in[0];  // [B,S,D]
  const float* lam  = (const float*)d_in[1];  // [S]
  const float* W    = (const float*)d_in[2];  // [D,4H]
  const float* U    = (const float*)d_in[3];  // [H,4H]
  const float* bias = (const float*)d_in[4];  // [4H]
  float* out = (float*)d_out;                 // [B,S,H] ++ [B,H] ++ [B,H]

  // workspace: x_proj f32 (256MB) + packed U hi/lo + packed W hi/lo (2MB)
  float* xp  = (float*)d_ws;
  u32*   uhi = (u32*)d_ws + XP_ELEMS;
  u32*   ulo = uhi + UPACK_DWORDS;
  u32*   whi = ulo + UPACK_DWORDS;
  u32*   wlo = whi + UPACK_DWORDS;

  pack_bmat_kernel<<<64, 256, 0, stream>>>(U, uhi, ulo);
  pack_bmat_kernel<<<64, 256, 0, stream>>>(W, whi, wlo);
  xproj_kernel<<<dim3(S_, B_ / 16, 1), 256, 0, stream>>>(x, whi, wlo, bias, xp);
  lstm_scan_kernel<<<4, 512, 0, stream>>>(xp, uhi, ulo, lam, out);
}
